// GCN_basic_35871566856596
// MI455X (gfx1250) — matrix operand order repaired and checked by reading, not compiled, not run on hardware
//
#include <hip/hip_runtime.h>
#include <hip/hip_bf16.h>

// ---------------------------------------------------------------------------
// GCN forward for MI455X (gfx1250, wave32, WMMA).
//   h = relu(spmm(x@W1)+b1); h = relu(spmm(h@W2)+b2); h = relu(spmm(h@W3)+b3)
//   out = h@Wd + bd
// GEMMs use v_wmma_f32_16x16x32_bf16 with 3-term bf16-split fp32 emulation.
// SpMM uses one wave32 per edge + global_atomic_add_f32 scatter.
// ---------------------------------------------------------------------------

typedef __bf16 bf16;
typedef __attribute__((ext_vector_type(16))) __bf16 v16bf;
typedef __attribute__((ext_vector_type(8)))  __bf16 v8bf;
typedef __attribute__((ext_vector_type(8)))  float  v8f;

#define DFEAT 256

// ---------------- helpers ----------------

__device__ __forceinline__ bf16 f2bf(float f) { return (bf16)f; }

// A-fragment loader: 16-bit A 16x32, contiguous k order (k = 16*half + s), matching B.
// lane L (half = L>=16): row m0 + L%16;
//   vector elems 0..7  = K k0 + 16*half .. +7
//   vector elems 8..15 = K k0 + 16*half + 8 .. +7
__device__ __forceinline__ v16bf load_a_frag(const bf16* __restrict__ row, int half) {
  v8bf r0 = *(const v8bf*)(row + half * 16);
  v8bf r1 = *(const v8bf*)(row + half * 16 + 8);
  v16bf a;
#pragma unroll
  for (int i = 0; i < 8; ++i) { a[i] = r0[i]; a[8 + i] = r1[i]; }
  return a;
}

// ---------------- setup kernels ----------------

// Build transposed, bf16-split weights: Wt[n*256+k] = W[k*ncolsW + n] (n<ncolsW else 0)
__global__ void k_wt_split(const float* __restrict__ W, bf16* __restrict__ Wth,
                           bf16* __restrict__ Wtl, int ncolsW, int npad) {
  int idx = blockIdx.x * blockDim.x + threadIdx.x;
  if (idx >= npad * DFEAT) return;
  int n = idx >> 8, k = idx & (DFEAT - 1);
  float v = (n < ncolsW) ? W[(size_t)k * ncolsW + n] : 0.0f;
  bf16 h = f2bf(v);
  Wth[idx] = h;
  Wtl[idx] = f2bf(v - (float)h);
}

// fp32 -> (hi, lo) bf16 split
__global__ void k_split(const float* __restrict__ src, bf16* __restrict__ hi,
                        bf16* __restrict__ lo, long n) {
  long idx = (long)blockIdx.x * blockDim.x + threadIdx.x;
  if (idx >= n) return;
  float v = src[idx];
  bf16 h = f2bf(v);
  hi[idx] = h;
  lo[idx] = f2bf(v - (float)h);
}

// relu(agg + b[col]) -> (hi, lo) bf16 split
__global__ void k_bias_relu_split(const float* __restrict__ agg, const float* __restrict__ b,
                                  bf16* __restrict__ hi, bf16* __restrict__ lo, long n) {
  long idx = (long)blockIdx.x * blockDim.x + threadIdx.x;
  if (idx >= n) return;
  float v = agg[idx] + b[idx & (DFEAT - 1)];
  v = fmaxf(v, 0.0f);
  bf16 h = f2bf(v);
  hi[idx] = h;
  lo[idx] = f2bf(v - (float)h);
}

__global__ void k_zero(float* __restrict__ p, long n) {
  long idx = (long)blockIdx.x * blockDim.x + threadIdx.x;
  if (idx < n) p[idx] = 0.0f;
}

// ---------------- WMMA GEMM ----------------
// out[nrow x ncol_valid] (row stride out_stride) = A[nrow x 256] @ B + optional bias.
// A given as bf16 split (Ah, Al), B given transposed+split (Bh/Bl: [npad x 256]).
// One 16x16 C tile per wave; 4 waves per block; grid.y = column tiles.
__global__ void __launch_bounds__(128)
k_gemm_wmma(const bf16* __restrict__ Ah, const bf16* __restrict__ Al,
            const bf16* __restrict__ Bh, const bf16* __restrict__ Bl,
            float* __restrict__ out, int nrow, int out_stride, int ncol_valid,
            const float* __restrict__ bias) {
  const int wave = threadIdx.x >> 5;
  const int lane = threadIdx.x & 31;
  const int mtile = blockIdx.x * 4 + wave;
  if (mtile * 16 >= nrow) return;          // wave-uniform: EXEC stays all-1s
  const int m0 = mtile * 16;
  const int n0 = blockIdx.y * 16;
  const int lm = lane & 15;
  const int half = lane >> 4;

  const bf16* arh = Ah + (size_t)(m0 + lm) * DFEAT;
  const bf16* arl = Al + (size_t)(m0 + lm) * DFEAT;
  const bf16* brh = Bh + (size_t)(n0 + lm) * DFEAT;
  const bf16* brl = Bl + (size_t)(n0 + lm) * DFEAT;

  v8f acc = {};
#pragma unroll
  for (int k0 = 0; k0 < DFEAT; k0 += 32) {
    v16bf ah = load_a_frag(arh + k0, half);
    v16bf al = load_a_frag(arl + k0, half);
    // B fragment (32x16): lane = col n0+lm, K = k0 + 16*half .. +15, contiguous in Wt row
    v16bf bh = *(const v16bf*)(brh + k0 + half * 16);
    v16bf bl = *(const v16bf*)(brl + k0 + half * 16);
    // fp32-emulating 3-product accumulation
    acc = __builtin_amdgcn_wmma_f32_16x16x32_bf16(false, ah, false, bh, (short)0, acc, false, false);
    acc = __builtin_amdgcn_wmma_f32_16x16x32_bf16(false, ah, false, bl, (short)0, acc, false, false);
    acc = __builtin_amdgcn_wmma_f32_16x16x32_bf16(false, al, false, bh, (short)0, acc, false, false);
  }

  // C/D layout: VGPR r, lanes 0-15 -> M=r, lanes 16-31 -> M=r+8; N = lane%16
  const int col = n0 + lm;
  if (col < ncol_valid) {
    const float bv = bias ? bias[col] : 0.0f;
#pragma unroll
    for (int r = 0; r < 8; ++r)
      out[(size_t)(m0 + r + half * 8) * out_stride + col] = acc[r] + bv;
  }
}

// ---------------- SpMM: agg[dst] += w * t[src], one wave per edge ----------------
__global__ void __launch_bounds__(256)
k_spmm(const float* __restrict__ t, const int* __restrict__ esrc,
       const int* __restrict__ edst, const float* __restrict__ ew,
       float* __restrict__ agg, int nE) {
  long gtid = (long)blockIdx.x * blockDim.x + threadIdx.x;
  int e = (int)(gtid >> 5);
  if (e >= nE) return;
  int lane = (int)(gtid & 31);
  int s = esrc[e], d = edst[e];
  float w = ew[e];
  const float4* in = (const float4*)(t + (size_t)s * DFEAT);
  float* outr = agg + (size_t)d * DFEAT;
  float4 v0 = in[lane];
  float4 v1 = in[lane + 32];
  int o = lane * 4;
  unsafeAtomicAdd(outr + o + 0, w * v0.x);
  unsafeAtomicAdd(outr + o + 1, w * v0.y);
  unsafeAtomicAdd(outr + o + 2, w * v0.z);
  unsafeAtomicAdd(outr + o + 3, w * v0.w);
  unsafeAtomicAdd(outr + 128 + o + 0, w * v1.x);
  unsafeAtomicAdd(outr + 128 + o + 1, w * v1.y);
  unsafeAtomicAdd(outr + 128 + o + 2, w * v1.z);
  unsafeAtomicAdd(outr + 128 + o + 3, w * v1.w);
}

// ---------------- launcher ----------------

static inline int cdiv(long a, long b) { return (int)((a + b - 1) / b); }

extern "C" void kernel_launch(void* const* d_in, const int* in_sizes, int n_in,
                              void* d_out, int out_size, void* d_ws, size_t ws_size,
                              hipStream_t stream) {
  const float* x    = (const float*)d_in[0];
  const int*   esrc = (const int*)d_in[1];
  const int*   edst = (const int*)d_in[2];
  const float* ew   = (const float*)d_in[3];
  const float* W1   = (const float*)d_in[4];
  const float* b1   = (const float*)d_in[5];
  const float* W2   = (const float*)d_in[6];
  const float* b2   = (const float*)d_in[7];
  const float* W3   = (const float*)d_in[8];
  const float* b3   = (const float*)d_in[9];
  const float* Wd   = (const float*)d_in[10];
  const float* bd   = (const float*)d_in[11];

  const int N  = in_sizes[0] / DFEAT;   // 100000
  const int nE = in_sizes[1];           // 3200000
  const int NCLS = in_sizes[11];        // 40
  const int NPAD = 48;                  // 3 tiles of 16

  // workspace carve-up (256B aligned)
  char* p = (char*)d_ws;
  size_t off = 0;
  auto carve = [&](size_t bytes) -> void* {
    void* r = p + off;
    off = (off + bytes + 255) & ~(size_t)255;
    return r;
  };
  bf16* wth[3]; bf16* wtl[3];
  for (int i = 0; i < 3; ++i) {
    wth[i] = (bf16*)carve((size_t)DFEAT * DFEAT * sizeof(bf16));
    wtl[i] = (bf16*)carve((size_t)DFEAT * DFEAT * sizeof(bf16));
  }
  bf16* wdth = (bf16*)carve((size_t)NPAD * DFEAT * sizeof(bf16));
  bf16* wdtl = (bf16*)carve((size_t)NPAD * DFEAT * sizeof(bf16));
  const long nElem = (long)N * DFEAT;
  bf16*  xh  = (bf16*)carve((size_t)nElem * sizeof(bf16));
  bf16*  xl  = (bf16*)carve((size_t)nElem * sizeof(bf16));
  float* t   = (float*)carve((size_t)nElem * sizeof(float));
  float* agg = (float*)carve((size_t)nElem * sizeof(float));

  const float* Ws[3] = {W1, W2, W3};
  const float* bs[3] = {b1, b2, b3};

  // 1) transpose+split weights
  for (int i = 0; i < 3; ++i)
    k_wt_split<<<cdiv((long)DFEAT * DFEAT, 256), 256, 0, stream>>>(Ws[i], wth[i], wtl[i], DFEAT, DFEAT);
  k_wt_split<<<cdiv((long)NPAD * DFEAT, 256), 256, 0, stream>>>(Wd, wdth, wdtl, NCLS, NPAD);

  // 2) split input features
  k_split<<<cdiv(nElem, 256), 256, 0, stream>>>(x, xh, xl, nElem);

  const dim3 gemmGrid(cdiv((long)N / 16, 4), DFEAT / 16);
  const dim3 headGrid(cdiv((long)N / 16, 4), NPAD / 16);
  const int spmmBlocks = cdiv((long)nE * 32, 256);

  // 3) three graph-conv layers
  for (int i = 0; i < 3; ++i) {
    k_gemm_wmma<<<gemmGrid, 128, 0, stream>>>(xh, xl, wth[i], wtl[i], t, N, DFEAT, DFEAT, nullptr);
    k_zero<<<cdiv(nElem, 256), 256, 0, stream>>>(agg, nElem);
    k_spmm<<<spmmBlocks, 256, 0, stream>>>(t, esrc, edst, ew, agg, nE);
    k_bias_relu_split<<<cdiv(nElem, 256), 256, 0, stream>>>(agg, bs[i], xh, xl, nElem);
  }

  // 4) dense head: out = h @ Wd + bd
  k_gemm_wmma<<<headGrid, 128, 0, stream>>>(xh, xl, wdth, wdtl, (float*)d_out, N, NCLS, NCLS, bd);
}
